// BilinearScorer_214748365180
// MI455X (gfx1250) — matrix operand order repaired and checked by reading, not compiled, not run on hardware
//
#include <hip/hip_runtime.h>
#include <hip/hip_bf16.h>
#include <stdint.h>

// ---------------------------------------------------------------------------
// BilinearScorer on MI455X (gfx1250, wave32, WMMA):
//   out[n,r] = sum_{h,k} pred[n,h] * U[h,r,k] * args[n,k]
//            + sum_k bias1[r*512+k] * args[n,k] + bias2[r]
// One GEMM:  out = P @ Ub,  P[n,(h,k)] = pred[n,h]*args[n,k] (rank-1, in regs),
// Ub[(h,k),r] = U[h,r,k] (f16, bias1 appended as h==512 row).
// M=4096, N=64, K=513*512. WMMA f32_16x16x32_f16, async-LDS staged U.
// ---------------------------------------------------------------------------

#define HID     512
#define ROLES   64
#define NTOK    4096
#define HTOT    513                      // 512 h rows + 1 bias1 row
#define NSLICE  8                        // K-split across workgroups
#define UB_ELEMS ((size_t)HTOT * ROLES * HID)          // 16,809,984 f16
#define UB_BYTES (UB_ELEMS * 2)                        // 33,619,968 B
#define H_ROW_BYTES (ROLES * HID * 2)                  // 65536 B per h in Ub
#define LDS_ROW_BYTES 1040                             // 512*2 + 16 pad (bank spread)
#define TILE_BYTES (ROLES * LDS_ROW_BYTES)             // 66,560 B per h tile
#define SMEM_BYTES (2 * TILE_BYTES)                    // double buffered: 133,120 B

typedef __attribute__((ext_vector_type(16))) _Float16 v16h;
typedef __attribute__((ext_vector_type(8)))  float    v8f;
typedef __attribute__((ext_vector_type(4)))  int      v4i;

typedef __attribute__((address_space(1))) v4i g_v4i;
typedef __attribute__((address_space(3))) v4i l_v4i;

static __device__ __forceinline__ g_v4i* to_glb(const void* p) {
  return (g_v4i*)(uint64_t)p;                    // inttoptr into AS1
}
static __device__ __forceinline__ l_v4i* to_lds(void* p) {
  // LDS generic addresses carry the byte offset in the low 32 bits.
  return (l_v4i*)(uint32_t)(uint64_t)p;          // inttoptr into AS3
}

#if defined(__HIP_DEVICE_COMPILE__) && __has_builtin(__builtin_amdgcn_global_load_async_to_lds_b128)
#define HAVE_ASYNC_LDS 1
#else
#define HAVE_ASYNC_LDS 0
#endif

static __device__ __forceinline__ void async_wait0() {
#if defined(__HIP_DEVICE_COMPILE__)
#if __has_builtin(__builtin_amdgcn_s_wait_asynccnt)
  __builtin_amdgcn_s_wait_asynccnt(0);
#else
  asm volatile("s_wait_asynccnt 0" ::: "memory");
#endif
#endif
}

// Stage Ub[h] (64 rows x 1024B) into padded LDS tile (rows of 1040B).
// 256 threads x 16 iters x 16B = 65536B.
static __device__ __forceinline__ void stage_tile(char* dst, const char* ub_h, int tid) {
#pragma unroll
  for (int i = 0; i < 16; ++i) {
    int idx = i * 256 + tid;
    int r = idx >> 6;        // 0..63
    int c = idx & 63;        // 0..63 (16B chunks)
    const char* s = ub_h + r * 1024 + c * 16;
    char* d = dst + r * LDS_ROW_BYTES + c * 16;
#if HAVE_ASYNC_LDS
    __builtin_amdgcn_global_load_async_to_lds_b128(to_glb(s), to_lds(d), 0, 0);
#else
    *(uint4*)d = *(const uint4*)s;
#endif
  }
}

// ---------------------------------------------------------------------------
// Kernel 1: convert U (f32) and bias1 (f32) into Ub (f16) in workspace.
// ---------------------------------------------------------------------------
__global__ __launch_bounds__(256) void cvt_u_kernel(const float* __restrict__ U,
                                                    const float* __restrict__ bias1,
                                                    _Float16* __restrict__ Ub) {
  size_t i = (size_t)blockIdx.x * 256 + threadIdx.x;
  if (i >= UB_ELEMS) return;
  const size_t ulen = (size_t)HID * ROLES * HID;   // 16,777,216
  float v = (i < ulen) ? U[i] : bias1[i - ulen];
  Ub[i] = (_Float16)v;
}

// ---------------------------------------------------------------------------
// Kernel 2: main WMMA GEMM.
// grid.x = 32 token groups (128 tokens), grid.y = 8 K-slices. 256 threads.
// Each wave: 16 tokens x all 64 roles (4 accumulators of v8f).
// ---------------------------------------------------------------------------
__global__ __launch_bounds__(256) void bilinear_wmma_kernel(
    const float* __restrict__ pred, const float* __restrict__ args,
    const _Float16* __restrict__ Ub, float* __restrict__ partials) {
  extern __shared__ char smem[];

  const int tid  = threadIdx.x;
  const int lane = tid & 31;
  const int wave = tid >> 5;          // 0..7
  const int m    = lane & 15;         // row within 16x16 tiles
  const int hi   = lane >> 4;         // 0/1: K-half selector (A/B layouts)

  const int sl = blockIdx.y;                       // K slice
  const int h0 = sl * 64;
  const int h1 = (sl == NSLICE - 1) ? HTOT : h0 + 64;
  const int n0 = blockIdx.x * 128 + wave * 16;     // wave's token base
  const int row = n0 + m;                          // this lane's token row

  // ---- args fragments: 16 tokens x 512 k, f32 -> f16, WMMA A layout.
  // Lane layout (16-bit A 16x32): lane<16 holds K {0..7,16..23}, lane>=16 holds
  // K {8..15,24..31} of each 32-chunk. 128 VGPRs, reused for this slice's 64 (last: 65) h-steps.
  const float* arow = args + (size_t)row * HID;
  v16h af[16];
#pragma unroll
  for (int kc = 0; kc < 16; ++kc) {
    const int kb = kc * 32 + hi * 8;
    float4 f0 = *(const float4*)(arow + kb);
    float4 f1 = *(const float4*)(arow + kb + 4);
    float4 f2 = *(const float4*)(arow + kb + 16);
    float4 f3 = *(const float4*)(arow + kb + 20);
    v16h t;
    t[0]  = (_Float16)f0.x; t[1]  = (_Float16)f0.y; t[2]  = (_Float16)f0.z; t[3]  = (_Float16)f0.w;
    t[4]  = (_Float16)f1.x; t[5]  = (_Float16)f1.y; t[6]  = (_Float16)f1.z; t[7]  = (_Float16)f1.w;
    t[8]  = (_Float16)f2.x; t[9]  = (_Float16)f2.y; t[10] = (_Float16)f2.z; t[11] = (_Float16)f2.w;
    t[12] = (_Float16)f3.x; t[13] = (_Float16)f3.y; t[14] = (_Float16)f3.z; t[15] = (_Float16)f3.w;
    af[kc] = t;
  }

  v8f acc[4];
#pragma unroll
  for (int rt = 0; rt < 4; ++rt) acc[rt] = 0.0f;

  const float* prow = pred + (size_t)row * HID;
  const char*  ubase = (const char*)Ub;

  // Prologue: stage first U tile.
  stage_tile(smem, ubase + (size_t)h0 * H_ROW_BYTES, tid);
  async_wait0();
  __syncthreads();

  union BU { uint4 q[2]; v16h v; };

  for (int h = h0; h < h1; ++h) {
    const char* cur = smem + ((h - h0) & 1) * TILE_BYTES;
    char* nxt = smem + (((h - h0) + 1) & 1) * TILE_BYTES;
    if (h + 1 < h1)
      stage_tile(nxt, ubase + (size_t)(h + 1) * H_ROW_BYTES, tid);

    // per-token scalar p = pred[row, h] (p = 1 for the appended bias1 row)
    float pf = (h < HID) ? prow[h] : 1.0f;
    _Float16 ph = (_Float16)pf;
    v16h pv;
#pragma unroll
    for (int e = 0; e < 16; ++e) pv[e] = ph;

    // Base of this lane's B-fragment rows inside the tile.
    const char* bbase = cur + m * LDS_ROW_BYTES + hi * 16;

    // B-fragment load for flat step s (kc = s>>2, rt = s&3): 2x ds_load_b128.
    auto loadB = [&](int s) {
      BU b;
      const char* bp = bbase + (s & 3) * (16 * LDS_ROW_BYTES) + (s >> 2) * 64;
      b.q[0] = *(const uint4*)bp;
      b.q[1] = *(const uint4*)(bp + 32);
      return b;
    };

    // Software pipeline, prefetch depth 2: rotating 3-slot B ring (no copies,
    // so regalloc cannot coalesce it to one buffer -> s_wait_dscnt relaxes and
    // two ds_load pairs stay in flight under each WMMA). A is double-buffered
    // by kc parity; its WMMA->VALU WAR v_nops co-execute with the XDL pipe.
    v16h a0 = pv * af[0];
    v16h a1;
    BU b[3];
    b[0] = loadB(0);
    b[1] = loadB(1);
#pragma unroll
    for (int s = 0; s < 64; ++s) {
      const int kc = s >> 2;
      const int rt = s & 3;
      if (s + 2 < 64) b[(s + 2) % 3] = loadB(s + 2);
      if (rt == 0 && kc < 15) {
        v16h& an = (kc & 1) ? a0 : a1;       // buffer for next kc
        an = pv * af[kc + 1];
      }
      const v16h& ac = (kc & 1) ? a1 : a0;
      acc[rt] = __builtin_amdgcn_wmma_f32_16x16x32_f16(
          false, ac, false, b[s % 3].v, (short)0, acc[rt], false, false);
    }

    async_wait0();
    __syncthreads();
  }

  // C/D layout: VGPR v, lanes 0-15 -> (M=v, N=lane); lanes 16-31 -> (M=v+8).
  float* pbase = partials + ((size_t)sl * NTOK + n0) * ROLES;
#pragma unroll
  for (int rt = 0; rt < 4; ++rt) {
#pragma unroll
    for (int v = 0; v < 8; ++v) {
      int tok = v + 8 * hi;
      pbase[(size_t)tok * ROLES + rt * 16 + m] = acc[rt][v];
    }
  }
}

// ---------------------------------------------------------------------------
// Kernel 3: reduce K-slice partials + bias2 (deterministic, no atomics).
// ---------------------------------------------------------------------------
__global__ __launch_bounds__(256) void reduce_kernel(const float* __restrict__ partials,
                                                     const float* __restrict__ bias2,
                                                     float* __restrict__ out) {
  int i = blockIdx.x * 256 + threadIdx.x;        // 0 .. 4096*64-1
  float s = bias2[i & (ROLES - 1)];
#pragma unroll
  for (int p = 0; p < NSLICE; ++p)
    s += partials[(size_t)p * (NTOK * ROLES) + i];
  out[i] = s;
}

extern "C" void kernel_launch(void* const* d_in, const int* in_sizes, int n_in,
                              void* d_out, int out_size, void* d_ws, size_t ws_size,
                              hipStream_t stream) {
  const float* pred  = (const float*)d_in[0];   // (8,512,512) f32
  const float* args  = (const float*)d_in[1];   // (8,512,512) f32
  const float* U     = (const float*)d_in[2];   // (512,64,512) f32
  const float* bias1 = (const float*)d_in[3];   // (1,32768) f32
  const float* bias2 = (const float*)d_in[4];   // (1,64) f32
  float* out = (float*)d_out;                   // (4096,64) f32

  _Float16* Ub       = (_Float16*)d_ws;
  float*    partials = (float*)((char*)d_ws + UB_BYTES);   // 8 MB

  // 1) U + bias1 -> f16 workspace (one-time, HBM-bound, ~100 MB traffic)
  cvt_u_kernel<<<(unsigned)((UB_ELEMS + 255) / 256), 256, 0, stream>>>(U, bias1, Ub);

  // 2) WMMA GEMM: 32 token-groups x 8 K-slices, 8 waves/WG, 130 KB LDS
  dim3 grid(NTOK / 128, NSLICE);
  bilinear_wmma_kernel<<<grid, 256, SMEM_BYTES, stream>>>(pred, args, Ub, partials);

  // 3) K-slice reduction + bias2
  reduce_kernel<<<(NTOK * ROLES) / 256, 256, 0, stream>>>(partials, bias2, out);
}
